// HopDistanceLoss_816043786445
// MI455X (gfx1250) — matrix operand order repaired and checked by reading, not compiled, not run on hardware
//
#include <hip/hip_runtime.h>

typedef __attribute__((ext_vector_type(16))) _Float16 v16h;
typedef __attribute__((ext_vector_type(4)))  _Float16 v4h;
typedef __attribute__((ext_vector_type(8)))  float    v8f;

#define B_  8
#define N_  4096
#define Q_  128
#define E_  32768

// ---------------------------------------------------------------------------
// Kernel 1: d_max = max(d_hw.max(), 1e-8); d16 = (f16)(d_hw / d_max)
// Single block of 1024 threads; 16384 elements.
// ---------------------------------------------------------------------------
__global__ void dnorm_kernel(const float* __restrict__ d_hw,
                             _Float16* __restrict__ d16) {
    __shared__ float red[1024];
    const int t = threadIdx.x;
    float m = 0.0f;
    for (int i = t; i < Q_ * Q_; i += 1024) m = fmaxf(m, d_hw[i]);
    red[t] = m;
    __syncthreads();
    for (int s = 512; s > 0; s >>= 1) {
        if (t < s) red[t] = fmaxf(red[t], red[t + s]);
        __syncthreads();
    }
    const float inv = 1.0f / fmaxf(red[0], 1e-8f);
    for (int i = t; i < Q_ * Q_; i += 1024)
        d16[i] = (_Float16)(d_hw[i] * inv);
}

// ---------------------------------------------------------------------------
// Kernel 2: S[m, q] = sum_r P[m, r] * d_norm[q, r]   (m = flattened b*N+n)
// One wave32 per 16-row tile. WMMA 16x16x32 f16 -> f32 accumulate.
//   A (16x32, f16): lane group g=lane/16, row m=lane%16,
//       half h -> K = K0 + 16g + h
//   B (32x16, f16): col n = lane%16, half h -> K = K0 + 16g + h
//       (contiguous 16 halves of d16 row n)
//   C/D (16x16, f32): VGPR v -> row v + 8g, col lane%16
// ---------------------------------------------------------------------------
__global__ void gemm_S_kernel(const float* __restrict__ P,
                              const _Float16* __restrict__ d16,
                              _Float16* __restrict__ S16) {
    const int wave = blockIdx.x * (blockDim.x >> 5) + (threadIdx.x >> 5);
    const int lane = threadIdx.x & 31;
    const int g    = lane >> 4;
    const int l16  = lane & 15;

    const long rowBase = (long)wave * 16;
    const float* __restrict__ Prow = P + (rowBase + l16) * Q_;

    v8f acc[8] = {};   // zero-initialized accumulators (8 N-tiles)

#pragma unroll
    for (int kc = 0; kc < 4; ++kc) {
        const int K0 = kc * 32;

        // ---- A fragment: four coalesced float4 loads + cvt to f16 ----
        const float4 f0 = *(const float4*)(Prow + K0 + 16 * g);
        const float4 f1 = *(const float4*)(Prow + K0 + 16 * g + 4);
        const float4 f2 = *(const float4*)(Prow + K0 + 16 * g + 8);
        const float4 f3 = *(const float4*)(Prow + K0 + 16 * g + 12);
        v16h a;
        a[0]  = (_Float16)f0.x; a[1]  = (_Float16)f0.y;
        a[2]  = (_Float16)f0.z; a[3]  = (_Float16)f0.w;
        a[4]  = (_Float16)f1.x; a[5]  = (_Float16)f1.y;
        a[6]  = (_Float16)f1.z; a[7]  = (_Float16)f1.w;
        a[8]  = (_Float16)f2.x; a[9]  = (_Float16)f2.y;
        a[10] = (_Float16)f2.z; a[11] = (_Float16)f2.w;
        a[12] = (_Float16)f3.x; a[13] = (_Float16)f3.y;
        a[14] = (_Float16)f3.z; a[15] = (_Float16)f3.w;

#pragma unroll
        for (int nt = 0; nt < 8; ++nt) {
            const int n = nt * 16 + l16;
            // 16 contiguous halves (32B, 32B-aligned) of d_norm row n
            const v16h b = *(const v16h*)(d16 + n * Q_ + K0 + 16 * g);
            acc[nt] = __builtin_amdgcn_wmma_f32_16x16x32_f16(
                false, a, false, b, (short)0, acc[nt], false, false);
        }
    }

    // ---- Store D: VGPR v -> row v + 8g, col nt*16 + l16 ----
#pragma unroll
    for (int nt = 0; nt < 8; ++nt) {
#pragma unroll
        for (int v = 0; v < 8; ++v) {
            const int m = v + 8 * g;
            S16[(rowBase + m) * Q_ + nt * 16 + l16] = (_Float16)acc[nt][v];
        }
    }
}

// ---------------------------------------------------------------------------
// Kernel 3: per-edge gathered dot product + mean reduction.
// One wave32 per edge; lane l handles elements 4l..4l+3.
// ---------------------------------------------------------------------------
__global__ void edge_dot_kernel(const float* __restrict__ P,
                                const _Float16* __restrict__ S16,
                                const int* __restrict__ edges,
                                float* __restrict__ out) {
    __shared__ float bsum;
    if (threadIdx.x == 0) bsum = 0.0f;
    __syncthreads();

    const int  wave = threadIdx.x >> 5;
    const int  lane = threadIdx.x & 31;
    const long idx  = (long)blockIdx.x * 8 + wave;       // edge id in [0, B*E)
    const int  b    = (int)(idx >> 15);                  // E = 2^15
    const int  e    = (int)(idx & (E_ - 1));

    const long ebase = ((long)b * E_ + e) * 2;
    const int  i = edges[ebase + 0];
    const int  j = edges[ebase + 1];

    const float*    Pi = P   + ((long)b * N_ + i) * Q_;
    const _Float16* Sj = S16 + ((long)b * N_ + j) * Q_;

    const float4 p = *(const float4*)(Pi + 4 * lane);
    const v4h    s = *(const v4h*)(Sj + 4 * lane);

    float sum = p.x * (float)s[0];
    sum = fmaf(p.y, (float)s[1], sum);
    sum = fmaf(p.z, (float)s[2], sum);
    sum = fmaf(p.w, (float)s[3], sum);

    // wave32 reduction
    for (int off = 16; off > 0; off >>= 1)
        sum += __shfl_xor(sum, off, 32);

    if (lane == 0) atomicAdd(&bsum, sum);
    __syncthreads();
    if (threadIdx.x == 0)
        atomicAdd(out, bsum * (1.0f / ((float)B_ * (float)E_)));
}

// ---------------------------------------------------------------------------
extern "C" void kernel_launch(void* const* d_in, const int* in_sizes, int n_in,
                              void* d_out, int out_size, void* d_ws, size_t ws_size,
                              hipStream_t stream) {
    (void)in_sizes; (void)n_in; (void)out_size; (void)ws_size;

    const float* P     = (const float*)d_in[0];
    const float* d_hw  = (const float*)d_in[1];
    const int*   edges = (const int*)d_in[2];
    float*       out   = (float*)d_out;

    _Float16* d16 = (_Float16*)d_ws;                                   // 32 KB
    _Float16* S16 = (_Float16*)((char*)d_ws + Q_ * Q_ * sizeof(_Float16)); // 8.4 MB

    hipMemsetAsync(d_out, 0, sizeof(float), stream);

    dnorm_kernel<<<1, 1024, 0, stream>>>(d_hw, d16);

    // (B*N)/16 = 2048 wave-tiles, 8 waves (256 threads) per block -> 256 blocks
    gemm_S_kernel<<<(B_ * N_ / 16) / 8, 256, 0, stream>>>(P, d16, S16);

    // B*E = 262144 edges, 8 waves per block -> 32768 blocks
    edge_dot_kernel<<<(B_ * E_) / 8, 256, 0, stream>>>(P, S16, edges, out);
}
